// CrossAttention_26946624815997
// MI455X (gfx1250) — matrix operand order repaired and checked by reading, not compiled, not run on hardware
//
#include <hip/hip_runtime.h>
#include <hip/hip_bf16.h>

typedef __attribute__((ext_vector_type(16))) _Float16 v16h;
typedef __attribute__((ext_vector_type(8)))  _Float16 v8h;
typedef __attribute__((ext_vector_type(8)))  float    v8f;

static constexpr int    kB   = 8;
static constexpr int    kC   = 128;
static constexpr int    kHW  = 16384;          // 128*128
static constexpr int    kNH  = 4;
static constexpr int    kHD  = 32;
static constexpr int    kX   = 256;            // patches per head
static constexpr int    kD   = 2048;           // hd*p*p
static constexpr size_t kTOT = (size_t)kB * kC * kHW; // 16,777,216

// ---------------------------------------------------------------------------
// BatchNorm statistics: one block per (tensor, channel). Reduces B*HW values.
// stats layout: [mean_rgb(128) | rstd_rgb(128) | mean_th(128) | rstd_th(128)]
// ---------------------------------------------------------------------------
__global__ __launch_bounds__(256) void bn_stats_kernel(
    const float* __restrict__ rgb, const float* __restrict__ th,
    float* __restrict__ stats) {
  int blk = blockIdx.x;                 // 0..255
  const float* x = (blk < kC) ? rgb : th;
  int c = blk & (kC - 1);
  int tid = threadIdx.x;
  float s = 0.f, s2 = 0.f;
  for (int b = 0; b < kB; ++b) {
    const float* p = x + ((size_t)b * kC + c) * kHW;
    for (int i = tid; i < kHW; i += 256) { float v = p[i]; s += v; s2 += v * v; }
  }
  __shared__ float sh[256], sh2[256];
  sh[tid] = s; sh2[tid] = s2; __syncthreads();
  for (int off = 128; off > 0; off >>= 1) {
    if (tid < off) { sh[tid] += sh[tid + off]; sh2[tid] += sh2[tid + off]; }
    __syncthreads();
  }
  if (tid == 0) {
    float n = (float)kB * (float)kHW;
    float mu = sh[0] / n;
    float var = sh2[0] / n - mu * mu;
    int base = (blk < kC) ? 0 : 2 * kC;
    stats[base + c] = mu;
    stats[base + kC + c] = rsqrtf(var + 1e-5f);
  }
}

// Apply BN and downconvert to f16.
__global__ __launch_bounds__(256) void bn_apply_kernel(
    const float* __restrict__ x, const float* __restrict__ stats,
    const float* __restrict__ g, const float* __restrict__ bb,
    _Float16* __restrict__ y) {
  size_t i = (size_t)blockIdx.x * 256 + threadIdx.x;
  if (i >= kTOT) return;
  int c = (int)((i >> 14) & (kC - 1));
  float v = (x[i] - stats[c]) * stats[kC + c] * g[c] + bb[c];
  y[i] = (_Float16)v;
}

__global__ __launch_bounds__(256) void cvt_f16_kernel(
    const float* __restrict__ x, _Float16* __restrict__ y, int n) {
  int i = blockIdx.x * 256 + threadIdx.x;
  if (i < n) y[i] = (_Float16)x[i];
}

// ---------------------------------------------------------------------------
// WMMA GEMM: D[m,n] = sum_k A[m,k]*B[k,n] (+bias[m]), f16 in / f32 accum.
// Block tile 128x64, BK=32, 256 threads = 8 waves. Each wave owns a 32x32
// output tile: 2 A-frags x 2 B-frags -> 4 v_wmma_f32_16x16x32_f16 per K step
// with fragment reuse. Double-buffered LDS: one barrier per K step, global
// staging of step k+1 overlaps WMMA on step k. Out-of-range rows/cols are
// handled by clamped (branch-free) staging; garbage rows/cols of A/B only
// affect garbage rows/cols of D, which the store-stage guards drop.
// B may be supplied transposed (btrans=1 -> B stored [N][K] row-major).
// Batched via blockIdx.z with element strides sA/sB/sO.
// ---------------------------------------------------------------------------
__global__ __launch_bounds__(256) void gemm_f16_wmma(
    const _Float16* __restrict__ A, int lda, long sA,
    const _Float16* __restrict__ B, int ldb, long sB, int btrans,
    float* __restrict__ out32, _Float16* __restrict__ out16, int ldo, long sO,
    const float* __restrict__ bias, int M, int N, int K) {
  __shared__ _Float16 As[2][128 * 32];   // [m][k]
  __shared__ _Float16 Bs[2][64 * 32];    // [n][k]  (transposed-in-LDS)

  long z = blockIdx.z;
  A += z * sA;
  B += z * sB;
  if (out32) out32 += z * sO;
  if (out16) out16 += z * sO;

  const int tid  = threadIdx.x;
  const int lane = tid & 31;
  const int wave = tid >> 5;
  const int mt   = (wave >> 1) * 32;     // wave row tile: 0/32/64/96
  const int nt   = (wave & 1) * 32;      // wave col tile: 0/32
  const int m0   = blockIdx.y * 128;
  const int n0   = blockIdx.x * 64;

  // --- branch-free staging coordinates (clamped into valid range) ---
  const int ar  = tid >> 1;              // A row 0..127
  const int akc = (tid & 1) * 16;        // A col base 0/16
  int arow = m0 + ar; if (arow > M - 1) arow = M - 1;
  const _Float16* aPtr = A + (size_t)arow * lda + akc;

  const int bkr = tid >> 3;              // 0..31 (row-major source k)
  int bnc = n0 + (tid & 7) * 8;          // col chunk (row-major source)
  if (bnc > N - 8) bnc = N - 8;
  const int bnr_l = tid >> 2;            // 0..63 (btrans source row)
  int bnr = n0 + bnr_l; if (bnr > N - 1) bnr = N - 1;
  const int bkc = (tid & 3) * 8;
  const _Float16* bPtr = btrans ? (B + (size_t)bnr * ldb + bkc)
                                : (B + (size_t)bkr * ldb + bnc);
  const size_t bStep = btrans ? (size_t)32 : (size_t)32 * ldb;
  const int bncl = (tid & 7) * 8;        // LDS col base for row-major source

  auto stage = [&](int buf) {
    v8h a0 = *(const v8h*)aPtr;
    v8h a1 = *(const v8h*)(aPtr + 8);
    __builtin_prefetch(aPtr + 32, 0, 3);
    *(v8h*)(&As[buf][ar * 32 + akc])     = a0;
    *(v8h*)(&As[buf][ar * 32 + akc + 8]) = a1;
    v8h bv = *(const v8h*)bPtr;
    __builtin_prefetch(bPtr + bStep, 0, 3);
    if (!btrans) {
#pragma unroll
      for (int j = 0; j < 8; ++j) Bs[buf][(bncl + j) * 32 + bkr] = bv[j];
    } else {
      *(v8h*)(&Bs[buf][bnr_l * 32 + bkc]) = bv;
    }
    aPtr += 32;
    bPtr += bStep;
  };

  v8f acc[2][2] = {{{}, {}}, {{}, {}}};

  stage(0);
  __syncthreads();
  int buf = 0;
  for (int k0 = 0; k0 < K; k0 += 32) {
    if (k0 + 32 < K) stage(buf ^ 1);

    const _Float16* asb = &As[buf][0];
    const _Float16* bsb = &Bs[buf][0];
    const int mrow = mt + (lane & 15);
    const int kb   = (lane & 16) ? 16 : 0;
    v16h af0, af1;
    {
      const _Float16* p = asb + mrow * 32 + kb;
      v8h lo = *(const v8h*)p;
      v8h hi = *(const v8h*)(p + 8);
#pragma unroll
      for (int i = 0; i < 8; ++i) { af0[i] = lo[i]; af0[i + 8] = hi[i]; }
      p += 16 * 32;
      lo = *(const v8h*)p;
      hi = *(const v8h*)(p + 8);
#pragma unroll
      for (int i = 0; i < 8; ++i) { af1[i] = lo[i]; af1[i + 8] = hi[i]; }
    }
    const int kb2 = (lane & 16) ? 16 : 0;
    v16h bf0 = *(const v16h*)(bsb + (nt + (lane & 15)) * 32 + kb2);
    v16h bf1 = *(const v16h*)(bsb + (nt + 16 + (lane & 15)) * 32 + kb2);

    acc[0][0] = __builtin_amdgcn_wmma_f32_16x16x32_f16(false, af0, false, bf0,
                                                       (short)0, acc[0][0], false, false);
    acc[0][1] = __builtin_amdgcn_wmma_f32_16x16x32_f16(false, af0, false, bf1,
                                                       (short)0, acc[0][1], false, false);
    acc[1][0] = __builtin_amdgcn_wmma_f32_16x16x32_f16(false, af1, false, bf0,
                                                       (short)0, acc[1][0], false, false);
    acc[1][1] = __builtin_amdgcn_wmma_f32_16x16x32_f16(false, af1, false, bf1,
                                                       (short)0, acc[1][1], false, false);
    __syncthreads();
    buf ^= 1;
  }

  // --- store (C/D layout: VGPR r -> m = r + 8*(lane>=16), n = lane&15) ---
#pragma unroll
  for (int s = 0; s < 2; ++s) {
#pragma unroll
    for (int t = 0; t < 2; ++t) {
      int n = n0 + nt + t * 16 + (lane & 15);
      if (n >= N) continue;
      int mbase = m0 + mt + s * 16 + ((lane & 16) ? 8 : 0);
#pragma unroll
      for (int r = 0; r < 8; ++r) {
        int m = mbase + r;
        if (m >= M) continue;
        float v = acc[s][t][r] + (bias ? bias[m] : 0.0f);
        if (out32) out32[(size_t)m * ldo + n] = v;
        if (out16) out16[(size_t)m * ldo + n] = (_Float16)v;
      }
    }
  }
}

// ---------------------------------------------------------------------------
// Patch encode: [B][C][HW] f16 -> mode 0: [B][NH][X][D]   (q, v)
//                                mode 1: [B][NH][D][X]   (k)
// d = hd*64 + ph*8 + pw ; x = nhp*16 + nwp ; h = nhp*8+ph ; w = nwp*8+pw
// ---------------------------------------------------------------------------
__global__ __launch_bounds__(256) void patch_encode_kernel(
    const _Float16* __restrict__ src, long srcBatchStride, int chanOff,
    _Float16* __restrict__ dst, int mode) {
  size_t idx = (size_t)blockIdx.x * 256 + threadIdx.x;
  if (idx >= kTOT) return;
  int d = (int)(idx & (kD - 1));
  int x = (int)((idx >> 11) & (kX - 1));
  int n = (int)((idx >> 19) & (kNH - 1));
  int b = (int)(idx >> 21);
  int c = chanOff + n * kHD + (d >> 6);
  int h = ((x >> 4) << 3) + ((d >> 3) & 7);
  int w = ((x & 15) << 3) + (d & 7);
  _Float16 v = src[(size_t)b * srcBatchStride + (size_t)c * kHW + h * 128 + w];
  size_t o = (mode == 0)
                 ? idx
                 : ((((size_t)b * kNH + n) * kD + d) * kX + x);
  dst[o] = v;
}

// Patch decode: [B][NH][X][D] -> [B][C][HW] f16
__global__ __launch_bounds__(256) void patch_decode_kernel(
    const _Float16* __restrict__ src, _Float16* __restrict__ dst) {
  size_t idx = (size_t)blockIdx.x * 256 + threadIdx.x;
  if (idx >= kTOT) return;
  int hw = (int)(idx & (kHW - 1));
  int c  = (int)((idx >> 14) & (kC - 1));
  int b  = (int)(idx >> 21);
  int n = c >> 5, hd = c & 31;
  int h = hw >> 7, w = hw & 127;
  int x = ((h >> 3) << 4) + (w >> 3);
  int d = (hd << 6) + ((h & 7) << 3) + (w & 7);
  dst[idx] = src[(((size_t)b * kNH + n) * kX + x) * kD + d];
}

// Row softmax with logit scale: f32 in -> f16 out, one block per row.
__global__ __launch_bounds__(128) void softmax_rows_kernel(
    const float* __restrict__ in, _Float16* __restrict__ out, int L, float scale) {
  int row = blockIdx.x;
  const float* p = in + (size_t)row * L;
  int tid = threadIdx.x;
  __shared__ float sh[128];
  float mx = -3.0e38f;
  for (int i = tid; i < L; i += 128) mx = fmaxf(mx, p[i] * scale);
  sh[tid] = mx; __syncthreads();
  for (int o = 64; o > 0; o >>= 1) {
    if (tid < o) sh[tid] = fmaxf(sh[tid], sh[tid + o]);
    __syncthreads();
  }
  mx = sh[0]; __syncthreads();
  float s = 0.f;
  for (int i = tid; i < L; i += 128) s += __expf(p[i] * scale - mx);
  sh[tid] = s; __syncthreads();
  for (int o = 64; o > 0; o >>= 1) {
    if (tid < o) sh[tid] += sh[tid + o];
    __syncthreads();
  }
  float inv = 1.0f / sh[0];
  for (int i = tid; i < L; i += 128)
    out[(size_t)row * L + i] = (_Float16)(__expf(p[i] * scale - mx) * inv);
}

// out[b, chanBase+c, hw] = x + sig(alpha)*sa + sig(beta)*ca   (f32 out, 2C chans)
__global__ __launch_bounds__(256) void combine_kernel(
    const float* __restrict__ x, const float* __restrict__ sa,
    const float* __restrict__ ca, const float* __restrict__ alpha,
    const float* __restrict__ beta, float* __restrict__ out, int chanBase) {
  size_t idx = (size_t)blockIdx.x * 256 + threadIdx.x;
  if (idx >= kTOT) return;
  int hw = (int)(idx & (kHW - 1));
  int c  = (int)((idx >> 14) & (kC - 1));
  int b  = (int)(idx >> 21);
  float sa_g = 1.0f / (1.0f + __expf(-alpha[0]));
  float ca_g = 1.0f / (1.0f + __expf(-beta[0]));
  out[((size_t)b * (2 * kC) + chanBase + c) * kHW + hw] =
      x[idx] + sa_g * sa[idx] + ca_g * ca[idx];
}

// ---------------------------------------------------------------------------
// Host orchestration
// ---------------------------------------------------------------------------
extern "C" void kernel_launch(void* const* d_in, const int* in_sizes, int n_in,
                              void* d_out, int out_size, void* d_ws, size_t ws_size,
                              hipStream_t stream) {
  (void)in_sizes; (void)n_in; (void)out_size; (void)ws_size;
  const float* rgb = (const float*)d_in[0];
  const float* th  = (const float*)d_in[1];
  const float* rgb_g = (const float*)d_in[2];
  const float* rgb_b = (const float*)d_in[3];
  const float* th_g  = (const float*)d_in[4];
  const float* th_b  = (const float*)d_in[5];
  float* out = (float*)d_out;

  // bump allocator over d_ws
  char* ws = (char*)d_ws;
  size_t off = 0;
  auto alloc = [&](size_t bytes) -> void* {
    void* p = ws + off;
    off = (off + bytes + 255) & ~(size_t)255;
    return p;
  };

  float*    stats  = (float*)alloc(4 * kC * sizeof(float));
  _Float16* w16q[4];
  _Float16* w16kv[4];
  _Float16* w16p[4];
  for (int br = 0; br < 4; ++br) {
    w16q[br]  = (_Float16*)alloc((size_t)kC * kC * 2);
    w16kv[br] = (_Float16*)alloc((size_t)2 * kC * kC * 2);
    w16p[br]  = (_Float16*)alloc((size_t)kC * kC * 2);
  }
  _Float16* nr16  = (_Float16*)alloc(kTOT * 2);
  _Float16* nt16  = (_Float16*)alloc(kTOT * 2);
  _Float16* q16   = (_Float16*)alloc(kTOT * 2);
  _Float16* kv16  = (_Float16*)alloc(kTOT * 4);       // 2C channels
  _Float16* encq  = (_Float16*)alloc(kTOT * 2);       // also k-buffer (channel)
  _Float16* enck  = (_Float16*)alloc(kTOT * 2);       // also qkv-out (spatial)
  _Float16* encv  = (_Float16*)alloc(kTOT * 2);       // also v-buffer (channel)
  float*    qk32  = (float*)alloc((size_t)kB * kNH * kX * kX * sizeof(float));
  _Float16* qksm  = (_Float16*)alloc((size_t)kB * kNH * kX * kX * 2);
  _Float16* att16 = (_Float16*)alloc(kTOT * 2);
  float*    sa32  = (float*)alloc(kTOT * sizeof(float));
  float*    ca32  = (float*)alloc(kTOT * sizeof(float));

  const int EB = (int)(kTOT / 256);   // elementwise blocks (65536)

  // 1) BN stats + apply (both modalities), activations -> f16
  bn_stats_kernel<<<2 * kC, 256, 0, stream>>>(rgb, th, stats);
  bn_apply_kernel<<<EB, 256, 0, stream>>>(rgb, stats, rgb_g, rgb_b, nr16);
  bn_apply_kernel<<<EB, 256, 0, stream>>>(th, stats + 2 * kC, th_g, th_b, nt16);

  // 2) weights -> f16 (branch order: sa_r, ca_r, sa_t, ca_t)
  for (int br = 0; br < 4; ++br) {
    const float* qw  = (const float*)d_in[10 + 6 * br + 0];
    const float* kvw = (const float*)d_in[10 + 6 * br + 2];
    const float* pw  = (const float*)d_in[10 + 6 * br + 4];
    cvt_f16_kernel<<<(kC * kC + 255) / 256, 256, 0, stream>>>(qw, w16q[br], kC * kC);
    cvt_f16_kernel<<<(2 * kC * kC + 255) / 256, 256, 0, stream>>>(kvw, w16kv[br], 2 * kC * kC);
    cvt_f16_kernel<<<(kC * kC + 255) / 256, 256, 0, stream>>>(pw, w16p[br], kC * kC);
  }

  auto gemm = [&](const _Float16* A, int lda, long sA,
                  const _Float16* B, int ldb, long sB, int btrans,
                  float* o32, _Float16* o16, int ldo, long sO,
                  const float* bias, int M, int N, int K, int batches) {
    dim3 g((N + 63) / 64, (M + 127) / 128, batches);
    gemm_f16_wmma<<<g, 256, 0, stream>>>(A, lda, sA, B, ldb, sB, btrans,
                                         o32, o16, ldo, sO, bias, M, N, K);
  };

  const long sBC  = (long)kC * kHW;       // per-batch stride, C channels
  const long sB2C = (long)2 * kC * kHW;   // per-batch stride, 2C channels
  const long sHead = (long)kHD * kHW;     // per-(b,head) stride in [B][C][HW]
  const long sXD  = (long)kX * kD;        // 524288
  const long sXX  = (long)kX * kX;        // 65536
  const float spScale = 0.022097086912079608f; // 1/sqrt(2048)
  const float chScale = 0.0078125f;            // 1/sqrt(16384)

  auto run_modality = [&](const float* xin, const _Float16* qin,
                          const _Float16* kvin, int saBr, int caBr,
                          const float* alpha, const float* beta, int chanBase) {
    const float* sa_qb  = (const float*)d_in[10 + 6 * saBr + 1];
    const float* sa_kvb = (const float*)d_in[10 + 6 * saBr + 3];
    const float* sa_pb  = (const float*)d_in[10 + 6 * saBr + 5];
    const float* ca_qb  = (const float*)d_in[10 + 6 * caBr + 1];
    const float* ca_kvb = (const float*)d_in[10 + 6 * caBr + 3];
    const float* ca_pb  = (const float*)d_in[10 + 6 * caBr + 5];

    // ---------------- spatial attention ----------------
    gemm(w16q[saBr], kC, 0, qin, kHW, sBC, 0,
         nullptr, q16, kHW, sBC, sa_qb, kC, kHW, kC, kB);
    gemm(w16kv[saBr], kC, 0, kvin, kHW, sBC, 0,
         nullptr, kv16, kHW, sB2C, sa_kvb, 2 * kC, kHW, kC, kB);

    patch_encode_kernel<<<EB, 256, 0, stream>>>(q16, sBC, 0, encq, 0);
    patch_encode_kernel<<<EB, 256, 0, stream>>>(kv16, sB2C, 0, enck, 1);
    patch_encode_kernel<<<EB, 256, 0, stream>>>(kv16, sB2C, kC, encv, 0);

    // qk[x,y] = sum_d q[x,d] * k[d,y]   (per b,head; K=2048)
    gemm(encq, kD, sXD, enck, kX, sXD, 0,
         qk32, nullptr, kX, sXX, nullptr, kX, kX, kD, kB * kNH);
    softmax_rows_kernel<<<kB * kNH * kX, 128, 0, stream>>>(qk32, qksm, kX, spScale);
    // qkv[x,d] = sum_y sm[x,y] * v[y,d]   (K=256) -> reuse enck as output
    gemm(qksm, kX, sXX, encv, kD, sXD, 0,
         nullptr, enck, kD, sXD, nullptr, kX, kD, kX, kB * kNH);

    patch_decode_kernel<<<EB, 256, 0, stream>>>(enck, att16);
    gemm(w16p[saBr], kC, 0, att16, kHW, sBC, 0,
         sa32, nullptr, kHW, sBC, sa_pb, kC, kHW, kC, kB);

    // ---------------- channel attention ----------------
    gemm(w16q[caBr], kC, 0, qin, kHW, sBC, 0,
         nullptr, q16, kHW, sBC, ca_qb, kC, kHW, kC, kB);
    // split kv conv so k/v land in uniform-stride buffers
    gemm(w16kv[caBr], kC, 0, kvin, kHW, sBC, 0,
         nullptr, encq, kHW, sBC, ca_kvb, kC, kHW, kC, kB);            // k
    gemm(w16kv[caBr] + (size_t)kC * kC, kC, 0, kvin, kHW, sBC, 0,
         nullptr, encv, kHW, sBC, ca_kvb + kC, kC, kHW, kC, kB);       // v

    // qk[i,j] = sum_x q[i,x] * k[j,x]  (btrans; K=16384, 32x32 out per b,head)
    gemm(q16, kHW, sHead, encq, kHW, sHead, 1,
         qk32, nullptr, kHD, (long)kHD * kHD, nullptr, kHD, kHD, kHW, kB * kNH);
    softmax_rows_kernel<<<kB * kNH * kHD, 128, 0, stream>>>(qk32, qksm, kHD, chScale);
    // qkv[i,x] = sum_j sm[i,j] * v[j,x]  (K=32) -> writes att16 channel slices
    gemm(qksm, kHD, (long)kHD * kHD, encv, kHW, sHead, 0,
         nullptr, att16, kHW, sHead, nullptr, kHD, kHW, kHD, kB * kNH);

    gemm(w16p[caBr], kC, 0, att16, kHW, sBC, 0,
         ca32, nullptr, kHW, sBC, ca_pb, kC, kHW, kC, kB);

    // ---------------- gated residual combine ----------------
    combine_kernel<<<EB, 256, 0, stream>>>(xin, sa32, ca32, alpha, beta, out, chanBase);
  };

  // rgb: spatial_attn(nr, nt) / channel_attn(nr, nt); branches sa_r=0, ca_r=1
  run_modality(rgb, nr16, nt16, 0, 1, (const float*)d_in[6], (const float*)d_in[7], 0);
  // thermal: spatial_attn(nt, nr) / channel_attn(nt, nr); branches sa_t=2, ca_t=3
  run_modality(th, nt16, nr16, 2, 3, (const float*)d_in[8], (const float*)d_in[9], kC);
}
